// MultiHeadAttention_10617159156318
// MI455X (gfx1250) — matrix operand order repaired and checked by reading, not compiled, not run on hardware
//
#include <hip/hip_runtime.h>

// MI455X (gfx1250) wave32 WMMA implementation.
// Dominant cost: z projections (294912x128 @ 128x128, twice) fused with score
// formation -> z (151MB) read once from HBM (~6.5us floor @23.3TB/s); bf16
// v_wmma_f32_16x16x32_bf16 makes the 19.3 GFLOP compute-free by comparison.
// fp32->bf16 on the streaming path uses v_perm_b32 packing (1.5 VALU/elem).

#define Bn 2
#define Nn 384
#define Mn 384
#define Qn 128
#define Hn 8
#define Dn 16

typedef __attribute__((ext_vector_type(16))) __bf16   v16bf;
typedef __attribute__((ext_vector_type(8)))  float    v8f;
typedef __attribute__((ext_vector_type(4)))  float    v4f;
typedef __attribute__((ext_vector_type(8)))  unsigned v8u;

static __device__ __forceinline__ unsigned short bf_bits(float f) {
  union { float f; unsigned u; } v; v.f = f;
  return (unsigned short)((v.u + 0x7FFFu + ((v.u >> 16) & 1u)) >> 16);  // RNE (cold paths)
}

// Pack two floats into two bf16 lanes with one v_perm_b32 (+2 rounding adds).
static __device__ __forceinline__ unsigned pack_bf16_pair(float f0, float f1) {
  unsigned u0 = __float_as_uint(f0) + 0x8000u;   // round-to-nearest, ties away
  unsigned u1 = __float_as_uint(f1) + 0x8000u;
  // v_perm_b32 {src0=u1(hi), src1=u0(lo)}: bytes [3,2]=u1.hi16, [1,0]=u0.hi16
  return __builtin_amdgcn_perm(u1, u0, 0x07060302u);
}

// A-matrix 16x32 bf16 fragment from fp32 row-major source.
// Order used here (contiguous k): lane<16 row=lane, elems {0..15}->K0..15;
// lane>=16 row=lane-16, K offset +16.  Caller passes p = row_base + kc*32 + half*16.
static __device__ __forceinline__ v16bf load_a_frag(const float* p) {
  v4f a0 = *(const v4f*)(p);
  v4f a1 = *(const v4f*)(p + 4);
  v4f b0 = *(const v4f*)(p + 8);
  v4f b1 = *(const v4f*)(p + 12);
  union { v16bf v; v8u u; } r;
  r.u[0] = pack_bf16_pair(a0[0], a0[1]);
  r.u[1] = pack_bf16_pair(a0[2], a0[3]);
  r.u[2] = pack_bf16_pair(a1[0], a1[1]);
  r.u[3] = pack_bf16_pair(a1[2], a1[3]);
  r.u[4] = pack_bf16_pair(b0[0], b0[1]);
  r.u[5] = pack_bf16_pair(b0[2], b0[3]);
  r.u[6] = pack_bf16_pair(b1[0], b1[1]);
  r.u[7] = pack_bf16_pair(b1[2], b1[3]);
  return r.v;
}

// Pre-staged bf16 B fragment: [kc][nt][lane][16 elems], 32B/lane, contiguous.
static __device__ __forceinline__ v16bf load_b_frag(const unsigned short* frags,
                                                    int kc, int nt, int lane) {
  return *(const v16bf*)(frags + (((kc * 8 + nt) * 32 + lane) << 4));
}

static __device__ __forceinline__ v8f wmma_bf16(v16bf a, v16bf b, v8f c) {
  return __builtin_amdgcn_wmma_f32_16x16x32_bf16(false, a, false, b, (short)0, c,
                                                 false, false);
}

// ---------------------------------------------------------------------------
// Kernel 0: swizzle 6 weight matrices (fp32 [H,Q,D]) into bf16 B-fragment
// layout. B layout: lane<16 holds K=kc*32+0..15 (elems 0..15), lane>=16 holds
// K=kc*32+16..31, column = nt*16 + (lane&15).
// w 0..4: B[k=q][col=h*16+d] ;  w==5 (theta2): B[k=h*16+d][col=q].
// ---------------------------------------------------------------------------
__global__ void stage_weights(const float* __restrict__ lx, const float* __restrict__ ly,
                              const float* __restrict__ lz1, const float* __restrict__ lz2,
                              const float* __restrict__ th1, const float* __restrict__ th2,
                              unsigned short* __restrict__ frags) {
  int gid = blockIdx.x * blockDim.x + threadIdx.x;
  if (gid >= 6 * 16384) return;
  int w    = gid >> 14;
  int r    = gid & 16383;
  int kc   = r >> 12;
  int nt   = (r >> 9) & 7;
  int lane = (r >> 4) & 31;
  int e    = r & 15;
  int half = lane >> 4;
  int l15  = lane & 15;
  int k = kc * 32 + half * 16 + e;   // K index
  int j = nt * 16 + l15;             // column index
  const float* W = (w == 0) ? lx : (w == 1) ? ly : (w == 2) ? lz1
                 : (w == 3) ? lz2 : (w == 4) ? th1 : th2;
  float val;
  if (w < 5) {                       // h = j>>4 (== nt), d = j&15, q = k
    val = W[(nt * Qn + k) * Dn + l15];
  } else {                           // theta2: h = k>>4, d = k&15, q = j
    val = W[((k >> 4) * Qn + j) * Dn + (k & 15)];
  }
  frags[w * 16384 + (((kc * 8 + nt) * 32 + lane) << 4) + e] = bf_bits(val);
}

// ---------------------------------------------------------------------------
// Generic WMMA GEMM: C[rows x 128] = A[rows x 128 fp32] @ staged bf16 W.
// One wave per 16-row tile, 8 accumulators (full 128 output cols).
// mode 0: fp32 row-major out (+ optional bias[128])
// mode 1: bf16 transposed t1T out: [b][h][d][n]  (row = b*Nn + n)
// ---------------------------------------------------------------------------
__global__ void wmma_proj_gemm(const float* __restrict__ A,
                               const unsigned short* __restrict__ frags,
                               const float* __restrict__ bias,
                               float* __restrict__ out_f32,
                               unsigned short* __restrict__ out_t1t,
                               int n_tiles, int mode) {
  int wid  = (blockIdx.x * blockDim.x + threadIdx.x) >> 5;
  int lane = threadIdx.x & 31;
  if (wid >= n_tiles) return;          // wave-uniform: EXEC stays all-ones
  int m0   = wid * 16;
  int half = lane >> 4;
  int l15  = lane & 15;

  v8f acc[8] = {};
  const float* abase = A + (long)(m0 + l15) * 128 + half * 16;
#pragma unroll
  for (int kc = 0; kc < 4; ++kc) {
    v16bf af = load_a_frag(abase + kc * 32);
#pragma unroll
    for (int nt = 0; nt < 8; ++nt) {
      acc[nt] = wmma_bf16(af, load_b_frag(frags, kc, nt, lane), acc[nt]);
    }
  }
  // C layout: lane<16 -> (M=v, N=lane); lane>=16 -> (M=v+8, N=lane-16)
  if (mode == 0) {
#pragma unroll
    for (int nt = 0; nt < 8; ++nt) {
      int col = nt * 16 + l15;
      float bv = bias ? bias[col] : 0.0f;
#pragma unroll
      for (int v = 0; v < 8; ++v) {
        int row = m0 + v + half * 8;
        out_f32[(long)row * 128 + col] = acc[nt][v] + bv;
      }
    }
  } else {
#pragma unroll
    for (int nt = 0; nt < 8; ++nt) {
#pragma unroll
      for (int v = 0; v < 8; ++v) {
        int row = m0 + v + half * 8;
        int b = row / Nn, n = row - b * Nn;
        out_t1t[((b * Hn + nt) * Dn + l15) * Nn + n] = bf_bits(acc[nt][v]);
      }
    }
  }
}

// ---------------------------------------------------------------------------
// Fused z kernel: per wave, one tile of 16 consecutive m rows for fixed (b,n).
// S1 = Z_tile @ Wz1, S2 = Z_tile @ Wz2 (16x128 each, bf16 WMMA), then
// score[h, m] = sum_d (S1+rx)(S2+ry) * D^-1/2, written to scores[b][h][m][n].
// ---------------------------------------------------------------------------
__global__ void fused_z_scores(const float* __restrict__ z,
                               const unsigned short* __restrict__ fz1,
                               const unsigned short* __restrict__ fz2,
                               const float* __restrict__ rx,
                               const float* __restrict__ ry,
                               float* __restrict__ scores) {
  int wid  = (blockIdx.x * blockDim.x + threadIdx.x) >> 5;
  int lane = threadIdx.x & 31;
  const int n_tiles = Bn * Nn * (Mn / 16);
  if (wid >= n_tiles) return;
  int b   = wid / (Nn * (Mn / 16));
  int rem = wid - b * (Nn * (Mn / 16));
  int n   = rem / (Mn / 16);
  int mt  = rem - n * (Mn / 16);
  int m0  = mt * 16;
  int half = lane >> 4;
  int l15  = lane & 15;

  long rowbase = ((long)(b * Nn + n) * Mn + m0);
  const float* abase = z + rowbase * Qn + (long)l15 * Qn + half * 16;

  v8f acc1[8] = {}, acc2[8] = {};
#pragma unroll
  for (int kc = 0; kc < 4; ++kc) {
    v16bf af = load_a_frag(abase + kc * 32);
    if (kc == 0 && mt < (Mn / 16) - 1)
      __builtin_prefetch(abase + 16 * Qn, 0, 0);   // next m-tile (global_prefetch_b8)
#pragma unroll
    for (int nt = 0; nt < 8; ++nt) {
      acc1[nt] = wmma_bf16(af, load_b_frag(fz1, kc, nt, lane), acc1[nt]);
      acc2[nt] = wmma_bf16(af, load_b_frag(fz2, kc, nt, lane), acc2[nt]);
    }
  }

  // Epilogue: column c = h*16 + l15 (h == nt exactly); rows split across halves.
  const float* rxrow = rx + (long)(b * Nn + n) * 128;
#pragma unroll
  for (int h = 0; h < 8; ++h) {
    int c = h * 16 + l15;
    float rxv = rxrow[c];
#pragma unroll
    for (int v = 0; v < 8; ++v) {
      int m = m0 + v + half * 8;
      float ryv = ry[(long)(b * Mn + m) * 128 + c];
      float p = (acc1[h][v] + rxv) * (acc2[h][v] + ryv);
      // sum over d = sum over the 16 lanes of each half-wave
      p += __shfl_xor(p, 1);
      p += __shfl_xor(p, 2);
      p += __shfl_xor(p, 4);
      p += __shfl_xor(p, 8);
      if (l15 == 0)
        scores[((long)(b * Hn + h) * Mn + m) * Nn + n] = p * 0.25f;  // D^-0.5
    }
  }
}

// ---------------------------------------------------------------------------
// Softmax over n (contiguous last axis of scores[b][h][m][n]); one wave/row.
// ---------------------------------------------------------------------------
__global__ void softmax_rows(float* __restrict__ s) {
  int row  = (blockIdx.x * blockDim.x + threadIdx.x) >> 5;
  int lane = threadIdx.x & 31;
  if (row >= Bn * Hn * Mn) return;
  float* p = s + (long)row * Nn;
  float mx = -3.402823466e38f;
#pragma unroll
  for (int j = 0; j < Nn / 32; ++j) mx = fmaxf(mx, p[lane + j * 32]);
  for (int o = 16; o > 0; o >>= 1) mx = fmaxf(mx, __shfl_xor(mx, o));
  float vals[Nn / 32];
  float sum = 0.0f;
#pragma unroll
  for (int j = 0; j < Nn / 32; ++j) {
    float e = __expf(p[lane + j * 32] - mx);
    vals[j] = e; sum += e;
  }
  for (int o = 16; o > 0; o >>= 1) sum += __shfl_xor(sum, o);
  float inv = 1.0f / sum;
#pragma unroll
  for (int j = 0; j < Nn / 32; ++j) p[lane + j * 32] = vals[j] * inv;
}

// ---------------------------------------------------------------------------
// r[b,m,h,:] = sum_n w[b,h,m,n] * t1[b,n,h,:]  + bias_theta.
// One wave per (b,h,m-tile); K = N = 384 (12 chunks), 16x16 output.
// B fragment read directly from bf16 t1T[b][h][d][n].
// ---------------------------------------------------------------------------
__global__ void attn_apply(const float* __restrict__ w,
                           const unsigned short* __restrict__ t1T,
                           const float* __restrict__ bias_theta,
                           float* __restrict__ rbuf) {
  int wid  = (blockIdx.x * blockDim.x + threadIdx.x) >> 5;
  int lane = threadIdx.x & 31;
  if (wid >= Bn * Hn * (Mn / 16)) return;
  int b   = wid / (Hn * (Mn / 16));
  int rem = wid - b * (Hn * (Mn / 16));
  int h   = rem / (Mn / 16);
  int mt  = rem - h * (Mn / 16);
  int m0  = mt * 16;
  int half = lane >> 4;
  int l15  = lane & 15;

  v8f acc = {};
  const float* abase = w + ((long)(b * Hn + h) * Mn + (m0 + l15)) * Nn + half * 16;
  const unsigned short* bbase =
      t1T + ((b * Hn + h) * Dn + l15) * Nn + half * 16;  // col=d=l15, K=n
#pragma unroll
  for (int kc = 0; kc < Nn / 32; ++kc) {
    v16bf af = load_a_frag(abase + kc * 32);             // fp32 weights -> bf16
    v16bf bf = *(const v16bf*)(bbase + kc * 32);         // 32B contiguous bf16
    acc = wmma_bf16(af, bf, acc);
  }
  float bt = bias_theta[h * Dn + l15];
#pragma unroll
  for (int v = 0; v < 8; ++v) {
    int m = m0 + v + half * 8;
    rbuf[(long)(b * Mn + m) * 128 + h * Dn + l15] = acc[v] + bt;
  }
}

// ---------------------------------------------------------------------------
extern "C" void kernel_launch(void* const* d_in, const int* in_sizes, int n_in,
                              void* d_out, int out_size, void* d_ws, size_t ws_size,
                              hipStream_t stream) {
  (void)in_sizes; (void)n_in; (void)out_size; (void)ws_size;
  const float* x   = (const float*)d_in[0];
  const float* y   = (const float*)d_in[1];
  const float* z   = (const float*)d_in[2];
  const float* lx  = (const float*)d_in[3];
  const float* ly  = (const float*)d_in[4];
  const float* lz1 = (const float*)d_in[5];
  const float* lz2 = (const float*)d_in[6];
  const float* blx = (const float*)d_in[7];
  const float* bly = (const float*)d_in[8];
  const float* th1 = (const float*)d_in[9];
  const float* th2 = (const float*)d_in[10];
  const float* bth = (const float*)d_in[11];
  float* out = (float*)d_out;
  char* ws = (char*)d_ws;

  // Workspace layout (bytes)
  unsigned short* frags  = (unsigned short*)(ws);            // 6*32768 = 196608
  float*          rx     = (float*)(ws + 196608);            // 768*128*4 = 393216
  float*          ry     = (float*)(ws + 589824);            // 393216
  unsigned short* t1T    = (unsigned short*)(ws + 983040);   // 2*8*16*384*2 = 196608
  float*          rbuf   = (float*)(ws + 1179648);           // 393216
  float*          scores = (float*)(ws + 1572864);           // 2*8*384*384*4 = 9437184

  // 0: stage all 6 weight matrices as bf16 B-fragments
  stage_weights<<<384, 256, 0, stream>>>(lx, ly, lz1, lz2, th1, th2, frags);
  // 1-3: r_x, r_y, t1 (transposed bf16) projections    (48 tiles each, 8 waves/blk)
  wmma_proj_gemm<<<6, 256, 0, stream>>>(x, frags + 0 * 16384, blx, rx,   nullptr, 48, 0);
  wmma_proj_gemm<<<6, 256, 0, stream>>>(y, frags + 1 * 16384, bly, ry,   nullptr, 48, 0);
  wmma_proj_gemm<<<6, 256, 0, stream>>>(x, frags + 4 * 16384, nullptr, nullptr, t1T, 48, 1);
  // 4: fused z projection + edge-conditioned scores    (18432 tiles)
  fused_z_scores<<<2304, 256, 0, stream>>>(z, frags + 2 * 16384, frags + 3 * 16384,
                                           rx, ry, scores);
  // 5: softmax over n (6144 rows, 8 waves/blk)
  softmax_rows<<<768, 256, 0, stream>>>(scores);
  // 6: attention apply (384 tiles)
  attn_apply<<<48, 256, 0, stream>>>(scores, t1T, bth, rbuf);
  // 7: final projection r @ theta2^T -> out [B,M,Q]
  wmma_proj_gemm<<<6, 256, 0, stream>>>(rbuf, frags + 5 * 16384, nullptr, out, nullptr, 48, 0);
}
